// MultiHeadAttention_79173427134478
// MI455X (gfx1250) — matrix operand order repaired and checked by reading, not compiled, not run on hardware
//
#include <hip/hip_runtime.h>
#include <hip/hip_bf16.h>

// ---------------------------------------------------------------------------
// MHA forward for MI455X (gfx1250, wave32, WMMA + Tensor Data Mover).
//   x:[B,S,1024] f32, 4 weights [1024,1024] f32 (nn.Linear: y = x @ W^T)
//   out:[B,S,1024] f32
//
// Fragment layouts (CDNA5 ISA 7.12.2), wave32, half = lane>>4, lr = lane&15:
//   A 16x32 (16-bit): lane row = lr; K chunks [half*8 .. +7] (VGPR0-3) and
//                     [16+half*8 .. +7] (VGPR4-7); packed pairs per VGPR.
//   B 32x16 (16-bit): lane col = lr; this code (A and B) puts half0 at K=0..15 (VGPR0-7, packed
//                     pairs (2v,2v+1)), half1 holds K=16..31.
//   C/D 16x16 (f32):  VGPR r -> row r + half*8 ; lane -> column.
// ---------------------------------------------------------------------------

#define D_MODEL 1024
#define NHEADS  16
#define HDIM    64
#define SEQ     2048

typedef __bf16 bf16_t;
typedef __attribute__((ext_vector_type(16))) __bf16 v16bf;
typedef __attribute__((ext_vector_type(8)))  float  v8f;
typedef unsigned int u32x4 __attribute__((ext_vector_type(4)));
typedef int          i32x4 __attribute__((ext_vector_type(4)));
typedef int          i32x8 __attribute__((ext_vector_type(8)));

#if __has_builtin(__builtin_amdgcn_tensor_load_to_lds)
#define HAVE_TDM 1
#else
#define HAVE_TDM 0
#endif

#if __has_builtin(__builtin_amdgcn_s_wait_tensorcnt)
#define WAIT_TENSORCNT(n) __builtin_amdgcn_s_wait_tensorcnt((short)(n))
#else
#define WAIT_TENSORCNT(n) asm volatile("s_wait_tensorcnt %0" ::"i"(n) : "memory")
#endif

union FragBF {
    v16bf    v;
    u32x4    q[2];
    unsigned d[8];
    __bf16   e[16];
};

union PairU {
    __bf16   h[2];
    unsigned u;
};

// A-fragment: rows contiguous over k (works for global AND LDS pointers).
__device__ __forceinline__ v16bf load_fragA_k32(const bf16_t* base, int row,
                                                int ld, int k0, int half) {
    FragBF f;
    const bf16_t* p = base + (size_t)row * ld + k0 + half * 16;
    f.q[0] = *(const u32x4*)(p);
    f.q[1] = *(const u32x4*)(p + 8);
    return f.v;
}

// B-fragment: columns contiguous over k in memory (column n at base + n*ld).
__device__ __forceinline__ v16bf load_fragB_k32(const bf16_t* base, int col,
                                                int ld, int k0, int half) {
    FragBF f;
    const bf16_t* p = base + (size_t)col * ld + k0 + half * 16;
    f.q[0] = *(const u32x4*)(p);
    f.q[1] = *(const u32x4*)(p + 8);
    return f.v;
}

__device__ __forceinline__ v8f wmma_bf16(v16bf a, v16bf b, v8f c) {
    return __builtin_amdgcn_wmma_f32_16x16x32_bf16(
        false, a, false, b, (short)0, c, false, false);
}

// ---------------------------------------------------------------------------
// Tensor Data Mover: async 2D tile load global->LDS (ISA 08, D# groups).
// data_size=2B; tile = tile_w x tile_h elements; row stride = stride elems.
// Unused dims: tile_dim2/3/4 = 0 ("unused"); far tensor dims set large so no
// OOB zeroing can trigger.  Wave-uniform arguments; tracked by TENSORcnt.
// ---------------------------------------------------------------------------
__device__ __forceinline__ void tdm_load_2d(unsigned lds_byte_off,
                                            const bf16_t* gptr,
                                            int tile_w, int tile_h, int stride) {
#if HAVE_TDM
    const unsigned long long ga = (unsigned long long)gptr;
    u32x4 g0;
    g0.x = 1u;                                              // count=1, user D#
    g0.y = lds_byte_off;                                    // lds_addr (bytes)
    g0.z = (unsigned)ga;                                    // global_addr lo
    g0.w = (unsigned)((ga >> 32) & 0x1FFFFFFu) | 0x80000000u; // addr hi | type=2
    i32x8 g1;
    g1[0] = 1 << 16;                          // data_size = 1 -> 2 bytes
    g1[1] = (tile_w & 0xFFFF) << 16;          // tensor_dim0 = tile_w
    g1[2] = 0;                                // dim0 hi | tensor_dim1 lo (dim1=1<<20)
    g1[3] = 16 | ((tile_w & 0xFFFF) << 16);   // tensor_dim1 hi=16 | tile_dim0
    g1[4] = tile_h & 0xFFFF;                  // tile_dim1 | tile_dim2=0
    g1[5] = stride;                           // tensor_dim0_stride lo32
    g1[6] = 0;                                // stride hi | dim1_stride lo
    g1[7] = 0;                                // dim1_stride hi
    i32x4 g2 = {1 << 20, 1 << 20, 0, 0};      // tensor_dim2/3 large; tile_dim3=0
    i32x4 g3 = {0, 0, 16, 0};                 // dim3_stride=0; tensor_dim4=1<<20
#if defined(__clang_major__) && (__clang_major__ >= 23)
    i32x8 g4 = {};
    __builtin_amdgcn_tensor_load_to_lds(g0, g1, g2, g3, g4, 0);
#else
    __builtin_amdgcn_tensor_load_to_lds(g0, g1, g2, g3, 0);
#endif
#else
    (void)lds_byte_off; (void)gptr; (void)tile_w; (void)tile_h; (void)stride;
#endif
}

// ---------------------------------------------------------------------------
// fp32 -> bf16 elementwise convert
// ---------------------------------------------------------------------------
__global__ void f32_to_bf16_kernel(const float* __restrict__ in,
                                   bf16_t* __restrict__ out, int n) {
    int i      = blockIdx.x * blockDim.x + threadIdx.x;
    int stride = gridDim.x * blockDim.x;
    for (; i < n; i += stride) out[i] = (bf16_t)in[i];
}

// ---------------------------------------------------------------------------
// GEMM: C[M,N] = scale * (A[M,K](bf16 row-major) @ W[N,K]^T).
// One wave computes a 32x64 tile: 2 A-frags x 4 B-frags -> 8 WMMAs / k-step.
// MODE 0: bf16 row-major (Q [scaled], K) | MODE 1: bf16 V^T | MODE 2: f32
// ---------------------------------------------------------------------------
template <int MODE>
__global__ __launch_bounds__(32) void gemm_bf16_32x64(
    const bf16_t* __restrict__ A, const bf16_t* __restrict__ W,
    void* __restrict__ outp, int M, int N, int K, int S, float scale) {
    const int lane = threadIdx.x;
    const int half = lane >> 4;
    const int lr   = lane & 15;

    const int ntiles = N >> 6;
    const int m0 = (blockIdx.x / ntiles) << 5;
    const int n0 = (blockIdx.x % ntiles) << 6;

    v8f acc[2][4] = {};

    const bf16_t* A0 = A + (size_t)m0 * K;
    const bf16_t* A1 = A + (size_t)(m0 + 16) * K;
    const bf16_t* W0 = W + (size_t)(n0 +  0) * K;
    const bf16_t* W1 = W + (size_t)(n0 + 16) * K;
    const bf16_t* W2 = W + (size_t)(n0 + 32) * K;
    const bf16_t* W3 = W + (size_t)(n0 + 48) * K;

    for (int k0 = 0; k0 < K; k0 += 32) {
        v16bf a0 = load_fragA_k32(A0, lr, K, k0, half);
        v16bf a1 = load_fragA_k32(A1, lr, K, k0, half);
        v16bf b0 = load_fragB_k32(W0, lr, K, k0, half);
        v16bf b1 = load_fragB_k32(W1, lr, K, k0, half);
        v16bf b2 = load_fragB_k32(W2, lr, K, k0, half);
        v16bf b3 = load_fragB_k32(W3, lr, K, k0, half);
        if (k0 + 32 < K) {   // stream-ahead hints -> global_prefetch_b8
            __builtin_prefetch(A0 + (size_t)lr * K + k0 + 32, 0, 1);
            __builtin_prefetch(W0 + (size_t)lr * K + k0 + 32, 0, 1);
        }
        acc[0][0] = wmma_bf16(a0, b0, acc[0][0]);
        acc[0][1] = wmma_bf16(a0, b1, acc[0][1]);
        acc[0][2] = wmma_bf16(a0, b2, acc[0][2]);
        acc[0][3] = wmma_bf16(a0, b3, acc[0][3]);
        acc[1][0] = wmma_bf16(a1, b0, acc[1][0]);
        acc[1][1] = wmma_bf16(a1, b1, acc[1][1]);
        acc[1][2] = wmma_bf16(a1, b2, acc[1][2]);
        acc[1][3] = wmma_bf16(a1, b3, acc[1][3]);
    }

#pragma unroll
    for (int mt = 0; mt < 2; ++mt) {
#pragma unroll
        for (int j = 0; j < 4; ++j) {
            const int n = n0 + j * 16 + lr;
            if (MODE == 1) {
                const int mb = m0 + mt * 16 + half * 8;
                const int b  = mb / S, s = mb - b * S;
                FragBF pk;
#pragma unroll
                for (int r = 0; r < 8; ++r)
                    pk.e[r] = (bf16_t)(acc[mt][j][r] * scale);
                *(u32x4*)((bf16_t*)outp + ((size_t)b * N + n) * S + s) = pk.q[0];
            } else {
#pragma unroll
                for (int r = 0; r < 8; ++r) {
                    const int m   = m0 + mt * 16 + r + half * 8;
                    const float v = acc[mt][j][r] * scale;
                    if (MODE == 0) ((bf16_t*)outp)[(size_t)m * N + n] = (bf16_t)v;
                    else           ((float*)outp)[(size_t)m * N + n]  = v;
                }
            }
        }
    }
}

// ---------------------------------------------------------------------------
// Causal flash attention, transposed formulation.
//   S^T = K_tile @ Q^T ; O^T += V^T_tile @ P^T
// K/V tiles staged in LDS by the TDM (double buffered), read as fragments.
// ktile: 32x64 bf16 (keys x d, ld 64); vtile: 64x32 bf16 (d x keys, ld 32).
// ---------------------------------------------------------------------------
template <bool MASKED>
__device__ __forceinline__ void attn_tile(
    const bf16_t* ktile, const bf16_t* vtile, int k0,
    v16bf qf0, v16bf qf1, int half, int lr, int q,
    v8f& o0, v8f& o1, v8f& o2, v8f& o3, float& m, float& l) {
    v16bf ka00 = load_fragA_k32(ktile,           lr, HDIM,  0, half);
    v16bf ka01 = load_fragA_k32(ktile,           lr, HDIM, 32, half);
    v16bf ka10 = load_fragA_k32(ktile + 16*HDIM, lr, HDIM,  0, half);
    v16bf ka11 = load_fragA_k32(ktile + 16*HDIM, lr, HDIM, 32, half);
    v16bf va0  = load_fragA_k32(vtile +  0 * 32, lr, 32, 0, half);
    v16bf va1  = load_fragA_k32(vtile + 16 * 32, lr, 32, 0, half);
    v16bf va2  = load_fragA_k32(vtile + 32 * 32, lr, 32, 0, half);
    v16bf va3  = load_fragA_k32(vtile + 48 * 32, lr, 32, 0, half);

    v8f st0 = {}, st1 = {};
    st0 = wmma_bf16(ka00, qf0, st0);
    st0 = wmma_bf16(ka01, qf1, st0);
    st1 = wmma_bf16(ka10, qf0, st1);
    st1 = wmma_bf16(ka11, qf1, st1);

    float vmax = -3.0e38f;
    float p0[8], p1[8];
#pragma unroll
    for (int r = 0; r < 8; ++r) {
        float s0 = st0[r];
        float s1 = st1[r];
        if (MASKED) {
            const int key0 = k0 + r + half * 8;
            s0 = (key0 > q)      ? -3.0e38f : s0;
            s1 = (key0 + 16 > q) ? -3.0e38f : s1;
        }
        p0[r] = s0;
        p1[r] = s1;
        vmax = fmaxf(vmax, fmaxf(s0, s1));
    }
    vmax = fmaxf(vmax, __shfl_xor(vmax, 16, 32));

    const float mnew = fmaxf(m, vmax);
    const float corr = __expf(m - mnew);
    m = mnew;

    float lsum = 0.0f;
    unsigned own0[4], own1[4];
#pragma unroll
    for (int v = 0; v < 4; ++v) {
        PairU u0, u1;
        const float e00 = __expf(p0[2 * v] - mnew);
        const float e01 = __expf(p0[2 * v + 1] - mnew);
        const float e10 = __expf(p1[2 * v] - mnew);
        const float e11 = __expf(p1[2 * v + 1] - mnew);
        lsum += (e00 + e01) + (e10 + e11);
        u0.h[0] = (bf16_t)e00; u0.h[1] = (bf16_t)e01;
        u1.h[0] = (bf16_t)e10; u1.h[1] = (bf16_t)e11;
        own0[v] = u0.u;
        own1[v] = u1.u;
    }
    l = l * corr + lsum;

    FragBF bp;
#pragma unroll
    for (int v = 0; v < 4; ++v) {
        const unsigned x0 = __shfl_xor(own0[v], 16, 32);
        const unsigned x1 = __shfl_xor(own1[v], 16, 32);
        bp.d[v]     = half ? x1 : own0[v];
        bp.d[4 + v] = half ? own1[v] : x0;
    }

#pragma unroll
    for (int i = 0; i < 8; ++i) {
        o0[i] *= corr; o1[i] *= corr; o2[i] *= corr; o3[i] *= corr;
    }
    o0 = wmma_bf16(va0, bp.v, o0);
    o1 = wmma_bf16(va1, bp.v, o1);
    o2 = wmma_bf16(va2, bp.v, o2);
    o3 = wmma_bf16(va3, bp.v, o3);
}

#define TILE_K_BYTES (32 * HDIM * 2)          // 4 KB
#define TILE_V_BYTES (HDIM * 32 * 2)          // 4 KB
#define BUF_BYTES    (TILE_K_BYTES + TILE_V_BYTES)

__global__ __launch_bounds__(32) void attn_fwd_kernel(
    const bf16_t* __restrict__ Qb, const bf16_t* __restrict__ Kb,
    const bf16_t* __restrict__ Vt, bf16_t* __restrict__ Ob, int S) {
    __shared__ __align__(16) bf16_t smem[2][BUF_BYTES / 2];   // ping-pong K|V

    const int lane = threadIdx.x;
    const int half = lane >> 4;
    const int lr   = lane & 15;

    const int qtiles = S >> 4;
    const int b   = blockIdx.x / (NHEADS * qtiles);
    const int rem = blockIdx.x % (NHEADS * qtiles);
    const int h   = rem / qtiles;
    const int q0  = (rem % qtiles) << 4;

    const bf16_t* Qbase = Qb + ((size_t)(b * S + q0)) * D_MODEL + h * HDIM;
    const bf16_t* Kbase = Kb + ((size_t)b * S) * D_MODEL + h * HDIM;
    const bf16_t* Vbase = Vt + ((size_t)b * D_MODEL + h * HDIM) * S;  // [64,S]

    const v16bf qf0 = load_fragB_k32(Qbase, lr, D_MODEL, 0, half);
    const v16bf qf1 = load_fragB_k32(Qbase, lr, D_MODEL, 32, half);

    v8f o0 = {}, o1 = {}, o2 = {}, o3 = {};
    float m = -3.0e38f;
    float l = 0.0f;
    const int q = q0 + lr;

    const int nfull = (q0 + 1) >> 5;           // tiles with k0+31 <= q0

#if HAVE_TDM
    // LDS byte offset of smem (AS(3) pointer value == LDS offset).
    const unsigned smem_off =
        (unsigned)(unsigned long long)(__attribute__((address_space(3))) char*)(char*)&smem[0][0];

    // issue tile 0 into buffer 0
    tdm_load_2d(smem_off, Kbase, HDIM, 32, D_MODEL);
    tdm_load_2d(smem_off + TILE_K_BYTES, Vbase, 32, HDIM, S);

    for (int kt = 0; kt < nfull; ++kt) {
        // issue tile kt+1 (may be the diagonal tile) into the other buffer
        const int nb = (kt + 1) & 1;
        const int nk0 = (kt + 1) << 5;
        tdm_load_2d(smem_off + nb * BUF_BYTES, Kbase + (size_t)nk0 * D_MODEL,
                    HDIM, 32, D_MODEL);
        tdm_load_2d(smem_off + nb * BUF_BYTES + TILE_K_BYTES, Vbase + nk0,
                    32, HDIM, S);
        WAIT_TENSORCNT(2);                      // tile kt's pair complete
        asm volatile("" ::: "memory");
        const bf16_t* kt_lds = &smem[kt & 1][0];
        attn_tile<false>(kt_lds, kt_lds + 32 * HDIM, kt << 5,
                         qf0, qf1, half, lr, q, o0, o1, o2, o3, m, l);
    }
    WAIT_TENSORCNT(0);
    asm volatile("" ::: "memory");
    const bf16_t* kt_lds = &smem[nfull & 1][0];
    attn_tile<true>(kt_lds, kt_lds + 32 * HDIM, nfull << 5,
                    qf0, qf1, half, lr, q, o0, o1, o2, o3, m, l);
#else
    // Fallback: stage tiles into LDS with plain loads/stores (one wave).
    for (int kt = 0; kt <= nfull; ++kt) {
        const int k0 = kt << 5;
        bf16_t* kt_lds = &smem[kt & 1][0];
        // lane i: K row i, V rows i and i+32: copy K tile (32x64) and V tile (64x32)
        for (int r = lane; r < 32; r += 32) {
#pragma unroll
            for (int c = 0; c < 4; ++c)
                *(u32x4*)(kt_lds + r * HDIM + c * 16) =
                    *(const u32x4*)(Kbase + (size_t)(k0 + r) * D_MODEL + c * 16);
        }
        for (int r = lane; r < HDIM; r += 32) {
#pragma unroll
            for (int c = 0; c < 2; ++c)
                *(u32x4*)(kt_lds + 32 * HDIM + r * 32 + c * 16) =
                    *(const u32x4*)(Vbase + (size_t)r * S + k0 + c * 16);
        }
        __syncthreads();   // host-parsable; single wave: orders LDS st->ld
        if (kt < nfull)
            attn_tile<false>(kt_lds, kt_lds + 32 * HDIM, k0,
                             qf0, qf1, half, lr, q, o0, o1, o2, o3, m, l);
        else
            attn_tile<true>(kt_lds, kt_lds + 32 * HDIM, k0,
                            qf0, qf1, half, lr, q, o0, o1, o2, o3, m, l);
    }
#endif

    const float ltot = l + __shfl_xor(l, 16, 32);
    const float inv  = 1.0f / ltot;

    bf16_t* obase = Ob + ((size_t)(b * S + q)) * D_MODEL + h * HDIM + half * 8;
    FragBF w0, w1, w2, w3;
#pragma unroll
    for (int r = 0; r < 8; ++r) {
        w0.e[r] = (bf16_t)(o0[r] * inv);
        w1.e[r] = (bf16_t)(o1[r] * inv);
        w2.e[r] = (bf16_t)(o2[r] * inv);
        w3.e[r] = (bf16_t)(o3[r] * inv);
    }
    *(u32x4*)(obase)      = w0.q[0];
    *(u32x4*)(obase + 16) = w1.q[0];
    *(u32x4*)(obase + 32) = w2.q[0];
    *(u32x4*)(obase + 48) = w3.q[0];
}

// ---------------------------------------------------------------------------
extern "C" void kernel_launch(void* const* d_in, const int* in_sizes, int n_in,
                              void* d_out, int out_size, void* d_ws, size_t ws_size,
                              hipStream_t stream) {
    const float* x  = (const float*)d_in[0];
    const float* qW = (const float*)d_in[1];
    const float* kW = (const float*)d_in[2];
    const float* vW = (const float*)d_in[3];
    const float* oW = (const float*)d_in[4];

    const int S  = SEQ;
    const int BS = in_sizes[0] / D_MODEL;    // B*S = 4096
    const int B  = BS / S;                   // 2

    char*  ws  = (char*)d_ws;
    size_t off = 0;
    bf16_t* Xb  = (bf16_t*)(ws + off); off += (size_t)BS * D_MODEL * sizeof(bf16_t);
    bf16_t* Wqb = (bf16_t*)(ws + off); off += (size_t)D_MODEL * D_MODEL * sizeof(bf16_t);
    bf16_t* Wkb = (bf16_t*)(ws + off); off += (size_t)D_MODEL * D_MODEL * sizeof(bf16_t);
    bf16_t* Wvb = (bf16_t*)(ws + off); off += (size_t)D_MODEL * D_MODEL * sizeof(bf16_t);
    bf16_t* Wob = (bf16_t*)(ws + off); off += (size_t)D_MODEL * D_MODEL * sizeof(bf16_t);
    bf16_t* Qb  = (bf16_t*)(ws + off); off += (size_t)BS * D_MODEL * sizeof(bf16_t);
    bf16_t* Kb  = (bf16_t*)(ws + off); off += (size_t)BS * D_MODEL * sizeof(bf16_t);
    bf16_t* Vt  = (bf16_t*)(ws + off); off += (size_t)BS * D_MODEL * sizeof(bf16_t);
    bf16_t* Ab  = (bf16_t*)(ws + off); off += (size_t)BS * D_MODEL * sizeof(bf16_t);

    {
        const int nx = BS * D_MODEL;
        f32_to_bf16_kernel<<<1024, 256, 0, stream>>>(x, Xb, nx);
        const int nw = D_MODEL * D_MODEL;
        f32_to_bf16_kernel<<<512, 256, 0, stream>>>(qW, Wqb, nw);
        f32_to_bf16_kernel<<<512, 256, 0, stream>>>(kW, Wkb, nw);
        f32_to_bf16_kernel<<<512, 256, 0, stream>>>(vW, Wvb, nw);
        f32_to_bf16_kernel<<<512, 256, 0, stream>>>(oW, Wob, nw);
    }

    const int gemm_grid = (BS >> 5) * (D_MODEL >> 6);   // 128 * 16 = 2048
    gemm_bf16_32x64<0><<<gemm_grid, 32, 0, stream>>>(Xb, Wqb, (void*)Qb,
                                                     BS, D_MODEL, D_MODEL, S, 0.125f);
    gemm_bf16_32x64<0><<<gemm_grid, 32, 0, stream>>>(Xb, Wkb, (void*)Kb,
                                                     BS, D_MODEL, D_MODEL, S, 1.0f);
    gemm_bf16_32x64<1><<<gemm_grid, 32, 0, stream>>>(Xb, Wvb, (void*)Vt,
                                                     BS, D_MODEL, D_MODEL, S, 1.0f);

    const int attn_grid = B * NHEADS * (S >> 4);        // 4096
    attn_fwd_kernel<<<attn_grid, 32, 0, stream>>>(Qb, Kb, Vt, Ab, S);

    gemm_bf16_32x64<2><<<gemm_grid, 32, 0, stream>>>(Ab, Wob, d_out,
                                                     BS, D_MODEL, D_MODEL, S, 1.0f);

    (void)n_in; (void)out_size; (void)ws_size;
}
